// SimpleMoEModel_91276644974696
// MI455X (gfx1250) — matrix operand order repaired and checked by reading, not compiled, not run on hardware
//
#include <hip/hip_runtime.h>
#include <hip/hip_bf16.h>
#include <stdint.h>

// ---------------- problem constants ----------------
static constexpr int B_  = 8;
static constexpr int S_  = 512;
static constexpr int H_  = 1024;
static constexpr int E_  = 8;
static constexpr int T_  = B_ * S_;      // 4096 tokens
static constexpr int CAP_ = T_ / E_;     // 512 capacity
static constexpr int NC_ = 8;            // S-chunks for mean reduction

// ---------------- WMMA types ----------------
typedef __attribute__((ext_vector_type(16))) __bf16 v16bf;
typedef __attribute__((ext_vector_type(8)))  float  v8f;

union Frag16 {
    uint32_t u[8];
    v16bf    v;
};

__device__ __forceinline__ uint32_t pack_bf16x2(float lo, float hi) {
#if __has_builtin(__builtin_amdgcn_cvt_pk_bf16_f32)
    typedef __attribute__((ext_vector_type(2))) __bf16 v2bf;
    union { v2bf v; uint32_t u; } cv;
    cv.v = __builtin_amdgcn_cvt_pk_bf16_f32(lo, hi);
    return cv.u;
#else
    uint32_t a = __float_as_uint(lo);
    uint32_t b = __float_as_uint(hi);
    // round-to-nearest-even bf16
    a += 0x7FFFu + ((a >> 16) & 1u);
    b += 0x7FFFu + ((b >> 16) & 1u);
    return (a >> 16) | (b & 0xFFFF0000u);
#endif
}

// =====================================================================
// GEMM: C[M,N] = A[M,K] @ W[K,N] + bias[N], f32 in/out, bf16 WMMA math.
// Block: 256 threads (8 waves). Block tile 128(M) x 128(N), K-step 32.
// Wave tile 32x64 = 2x4 fragments -> 8 v_wmma_f32_16x16x32_bf16 / step.
// Double-buffered LDS with COMPILE-TIME buffer indices (K-loop manually
// unrolled by 2) so all ds addresses are loop-invariant + immediates.
// Requires M%128==0, N%128==0, K%64==0 and K>=64 (here K=1024).
// =====================================================================
static constexpr int LSTR = 17;                  // u32 stride per row/col (+1 pad)
static constexpr int LB_OFF = 128 * LSTR;        // B region offset inside a buffer
static constexpr int LBUF = 2 * 128 * LSTR;      // u32 per buffer (A + B)

__launch_bounds__(256)
__global__ void gemm_bias_bf16(const float* __restrict__ A,
                               const float* __restrict__ W,
                               const float* __restrict__ bias,
                               float* __restrict__ C,
                               int M, int N, int K,
                               long long sA, long long sW,
                               long long sBias, long long sC)
{
    const int e = blockIdx.z;
    A    += (long long)e * sA;
    W    += (long long)e * sW;
    bias += (long long)e * sBias;
    C    += (long long)e * sC;

    __shared__ uint32_t lds[2][LBUF];            // ~34 KB total

    const int tid  = threadIdx.x;
    const int lane = tid & 31;
    const int wave = tid >> 5;
    const int wm   = wave & 3;       // 0..3 (M direction, 4 x 32 = 128)
    const int wn   = wave >> 2;      // 0..1 (N direction, 2 x 64 = 128)
    const int half = lane >> 4;      // 0/1 half-wave
    const int l15  = lane & 15;

    const int Mb = blockIdx.x * 128;
    const int Nb = blockIdx.y * 128;

    // ---- staging task mapping (2 tasks per thread for A and for B) ----
    const int a_r0  = tid >> 2;          // rows 0..63   (task i adds +64)
    const int a_kpg = tid & 3;           // kpair group 0..3
    const int b_kp0 = tid >> 5;          // 0..7  (task i adds +8)
    const int b_cg  = tid & 31;          // col group 0..31

    const v8f vzero = {0.f, 0.f, 0.f, 0.f, 0.f, 0.f, 0.f, 0.f};
    v8f acc[2][4];
    #pragma unroll
    for (int fm = 0; fm < 2; ++fm)
        #pragma unroll
        for (int fn = 0; fn < 4; ++fn) acc[fm][fn] = vzero;

    float4 fa[2][2];   // A staging regs: 2 tasks x 8 floats
    float4 fb[2][2];   // B staging regs: 2 tasks x (row k, row k+1) float4

    auto GLOAD = [&](int k0) {
        #pragma unroll
        for (int i = 0; i < 2; ++i) {
            const int r = a_r0 + i * 64;
            const float* pa = A + (long long)(Mb + r) * K + (k0 + 8 * a_kpg);
            fa[i][0] = *(const float4*)pa;
            fa[i][1] = *(const float4*)(pa + 4);
            const int kp = b_kp0 + i * 8;
            const float* pb = W + (long long)(k0 + 2 * kp) * N + (Nb + 4 * b_cg);
            fb[i][0] = *(const float4*)pb;
            fb[i][1] = *(const float4*)(pb + N);
        }
    };

    auto LSTORE = [&](int buf) {   // buf is a literal at every call site
        #pragma unroll
        for (int i = 0; i < 2; ++i) {
            const int r = a_r0 + i * 64;
            uint32_t* pa = &lds[buf][r * LSTR + 4 * a_kpg];
            pa[0] = pack_bf16x2(fa[i][0].x, fa[i][0].y);
            pa[1] = pack_bf16x2(fa[i][0].z, fa[i][0].w);
            pa[2] = pack_bf16x2(fa[i][1].x, fa[i][1].y);
            pa[3] = pack_bf16x2(fa[i][1].z, fa[i][1].w);
            const int kp = b_kp0 + i * 8;
            uint32_t* pb = &lds[buf][LB_OFF + (4 * b_cg) * LSTR + kp];
            pb[0 * LSTR] = pack_bf16x2(fb[i][0].x, fb[i][1].x);
            pb[1 * LSTR] = pack_bf16x2(fb[i][0].y, fb[i][1].y);
            pb[2 * LSTR] = pack_bf16x2(fb[i][0].z, fb[i][1].z);
            pb[3 * LSTR] = pack_bf16x2(fb[i][0].w, fb[i][1].w);
        }
    };

    auto COMPUTE = [&](int buf) {  // buf is a literal at every call site
        // A fragments (resident), contiguous k = 16*half + s layout
        Frag16 a[2];
        #pragma unroll
        for (int fm = 0; fm < 2; ++fm) {
            const int r = wm * 32 + fm * 16 + l15;
            const uint32_t* p = &lds[buf][r * LSTR];
            #pragma unroll
            for (int v = 0; v < 4; ++v) a[fm].u[v]     = p[half * 8 + v];
            #pragma unroll
            for (int v = 0; v < 4; ++v) a[fm].u[4 + v] = p[half * 8 + 4 + v];
        }
        // stream B fragments, 2 WMMAs each
        #pragma unroll
        for (int fn = 0; fn < 4; ++fn) {
            const int c = wn * 64 + fn * 16 + l15;
            const uint32_t* p = &lds[buf][LB_OFF + c * LSTR + half * 8];
            Frag16 b;
            #pragma unroll
            for (int v = 0; v < 8; ++v) b.u[v] = p[v];
            acc[0][fn] = __builtin_amdgcn_wmma_f32_16x16x32_bf16(
                false, a[0].v, false, b.v, (short)0, acc[0][fn], false, false);
            acc[1][fn] = __builtin_amdgcn_wmma_f32_16x16x32_bf16(
                false, a[1].v, false, b.v, (short)0, acc[1][fn], false, false);
        }
    };

    // ---- software pipeline, unrolled x2 (constant buffer indices) ----
    // Tiles: K/32 total (even). Prologue stages tile0 -> buf0 and holds
    // tile1 in regs; each loop iteration computes two tiles; epilogue
    // computes the last two.
    GLOAD(0);
    LSTORE(0);
    GLOAD(32);
    __syncthreads();
    for (int k0 = 64; k0 < K; k0 += 64) {   // k0 = 64 .. K-64
        COMPUTE(0);          // tile (k0-64)
        LSTORE(1);           // stage tile (k0-32) from regs
        __syncthreads();
        GLOAD(k0);           // prefetch tile k0 (hidden under COMPUTE(1))
        COMPUTE(1);          // tile (k0-32)
        LSTORE(0);           // stage tile k0
        __syncthreads();
        GLOAD(k0 + 32);      // prefetch tile k0+32 (hidden under next COMPUTE(0))
    }
    COMPUTE(0);              // tile K-64
    LSTORE(1);
    __syncthreads();
    COMPUTE(1);              // tile K-32

    // ---- epilogue: C layout m = v + 8*half, n = lane&15 ----
    #pragma unroll
    for (int fm = 0; fm < 2; ++fm)
        #pragma unroll
        for (int fn = 0; fn < 4; ++fn)
            #pragma unroll
            for (int v = 0; v < 8; ++v) {
                const int row = Mb + wm * 32 + fm * 16 + half * 8 + v;
                const int col = Nb + wn * 64 + fn * 16 + l15;
                C[(long long)row * N + col] = acc[fm][fn][v] + bias[col];
            }
}

// =====================================================================
// Gating: per-token logits = hidden[t] @ wg  (E=8), softmax + argmax.
// =====================================================================
__launch_bounds__(256)
__global__ void gating_kernel(const float* __restrict__ hidden,
                              const float* __restrict__ wg,
                              int* __restrict__ expert_idx,
                              float* __restrict__ gate_val)
{
    const int wave = threadIdx.x >> 5;
    const int lane = threadIdx.x & 31;
    const int t    = blockIdx.x * 8 + wave;

    const float* xr = hidden + (long long)t * H_;
    float acc[E_];
    #pragma unroll
    for (int e = 0; e < E_; ++e) acc[e] = 0.f;

    for (int i = 0; i < H_ / 32; ++i) {
        const int h = i * 32 + lane;
        const float xv = xr[h];
        const float* wr = wg + (long long)h * E_;
        #pragma unroll
        for (int e = 0; e < E_; ++e) acc[e] += xv * wr[e];
    }
    #pragma unroll
    for (int off = 16; off > 0; off >>= 1) {
        #pragma unroll
        for (int e = 0; e < E_; ++e) acc[e] += __shfl_xor(acc[e], off, 32);
    }
    if (lane == 0) {
        float m = acc[0]; int am = 0;
        #pragma unroll
        for (int e = 1; e < E_; ++e)
            if (acc[e] > m) { m = acc[e]; am = e; }   // strict > : first max
        float s = 0.f;
        #pragma unroll
        for (int e = 0; e < E_; ++e) s += __expf(acc[e] - m);
        expert_idx[t] = am;
        gate_val[t]   = 1.0f / s;  // softmax prob of the argmax expert
    }
}

// =====================================================================
// Capacity assignment: single-wave ballot prefix scan (token order).
// =====================================================================
__global__ void assign_kernel(const int* __restrict__ expert_idx,
                              const float* __restrict__ gate_val,
                              int* __restrict__ slot,
                              float* __restrict__ cw,
                              int* __restrict__ keep,
                              int* __restrict__ token_for_slot,
                              int* __restrict__ count)
{
    const int lane = threadIdx.x;               // 32 threads
    int base[E_];
    #pragma unroll
    for (int e = 0; e < E_; ++e) base[e] = 0;
    const unsigned lt = (1u << lane) - 1u;

    for (int chunk = 0; chunk < T_ / 32; ++chunk) {
        const int t = chunk * 32 + lane;
        const int e = expert_idx[t];
        int my_slot = 0;
        #pragma unroll
        for (int ex = 0; ex < E_; ++ex) {
            const unsigned m = (unsigned)__ballot(e == ex);
            if (e == ex) my_slot = base[ex] + __popc(m & lt);
            base[ex] += __popc(m);
        }
        const int kept = (my_slot < CAP_) ? 1 : 0;
        slot[t] = my_slot;
        keep[t] = kept;
        cw[t]   = kept ? gate_val[t] : 0.0f;
        if (kept) token_for_slot[e * CAP_ + my_slot] = t;
    }
    if (lane == 0) {
        #pragma unroll
        for (int ex = 0; ex < E_; ++ex)
            count[ex] = base[ex] < CAP_ ? base[ex] : CAP_;
    }
}

// =====================================================================
// Gather tokens into dispatch buffer xd[E*CAP, H].
// CDNA5 async global<->LDS DMA (ASYNCcnt): each 4KB row moves
// global -> LDS -> global without a VGPR round-trip.
// =====================================================================
__launch_bounds__(256)
__global__ void gather_kernel(const float* __restrict__ src,
                              const int* __restrict__ token_for_slot,
                              const int* __restrict__ count,
                              float* __restrict__ xd)
{
    __shared__ __align__(16) float smem[H_];     // 4 KB staging
    const int sl = blockIdx.x;                   // expert*CAP + c
    const int e  = sl / CAP_;
    const int c  = sl % CAP_;
    if (c >= count[e]) return;
    const int t = token_for_slot[sl];
    const float* s = src + (long long)t * H_ + threadIdx.x * 4;
    float*       d = xd  + (long long)sl * H_ + threadIdx.x * 4;
    const uint32_t loff = (uint32_t)(uintptr_t)(&smem[threadIdx.x * 4]);

    asm volatile("global_load_async_to_lds_b128 %0, %1, off"
                 :: "v"(loff), "v"(s) : "memory");
    asm volatile("s_wait_asynccnt 0x0" ::: "memory");
    asm volatile("global_store_async_from_lds_b128 %0, %1, off"
                 :: "v"(d), "v"(loff) : "memory");
    asm volatile("s_wait_asynccnt 0x0" ::: "memory");
}

// =====================================================================
// Combine/scatter: out[t] = cw[t] * h2[expert[t]*CAP + slot[t]] (or 0)
// =====================================================================
__launch_bounds__(256)
__global__ void scatter_kernel(const float* __restrict__ h2,
                               const int* __restrict__ expert_idx,
                               const int* __restrict__ slot,
                               const int* __restrict__ keep,
                               const float* __restrict__ cw,
                               float* __restrict__ out)
{
    const int t = blockIdx.x;
    float4* d = (float4*)(out + (long long)t * H_);
    if (keep[t]) {
        const long long r = (long long)(expert_idx[t] * CAP_ + slot[t]) * H_;
        const float4* s = (const float4*)(h2 + r);
        const float g = cw[t];
        float4 v = s[threadIdx.x];
        d[threadIdx.x] = make_float4(g * v.x, g * v.y, g * v.z, g * v.w);
    } else {
        d[threadIdx.x] = make_float4(0.f, 0.f, 0.f, 0.f);
    }
}

// =====================================================================
// Mean phase 1: partial[chunk][bh] = sum over S/NC s of (a + r)
// =====================================================================
__launch_bounds__(256)
__global__ void mean_part_kernel(const float* __restrict__ a,
                                 const float* __restrict__ r,
                                 float* __restrict__ partial)
{
    const int idx   = blockIdx.x * 256 + threadIdx.x;   // 0 .. B*H*NC-1
    const int bh    = idx % (B_ * H_);
    const int chunk = idx / (B_ * H_);
    const int bb    = bh / H_;
    const int h     = bh % H_;
    const int s0    = chunk * (S_ / NC_);
    const float* pa = a + ((long long)bb * S_ + s0) * H_ + h;
    const float* pr = r + ((long long)bb * S_ + s0) * H_ + h;
    float s = 0.f;
    for (int t = 0; t < S_ / NC_; ++t) {
        s += pa[(long long)t * H_] + pr[(long long)t * H_];
    }
    partial[idx] = s;
}

// =====================================================================
// Mean phase 2: sent[bh] = (1/S) * sum_chunk partial[chunk][bh]
// =====================================================================
__launch_bounds__(256)
__global__ void mean_comb_kernel(const float* __restrict__ partial,
                                 float* __restrict__ sent)
{
    const int bh = blockIdx.x * 256 + threadIdx.x;      // 0 .. B*H-1
    float s = 0.f;
    #pragma unroll
    for (int c = 0; c < NC_; ++c) s += partial[c * (B_ * H_) + bh];
    sent[bh] = s * (1.0f / S_);
}

// =====================================================================
// loss = -mean_b ( sent[b, y[b]] - logsumexp_h sent[b,h] )
// =====================================================================
__launch_bounds__(256)
__global__ void loss_kernel(const float* __restrict__ sent,
                            const int* __restrict__ y,
                            float* __restrict__ out)
{
    __shared__ float red[256];
    float loss = 0.f;
    for (int b = 0; b < B_; ++b) {
        const float* row = sent + (long long)b * H_;
        float m = -3.4e38f;
        for (int h = threadIdx.x; h < H_; h += 256) m = fmaxf(m, row[h]);
        red[threadIdx.x] = m; __syncthreads();
        for (int s = 128; s > 0; s >>= 1) {
            if (threadIdx.x < s)
                red[threadIdx.x] = fmaxf(red[threadIdx.x], red[threadIdx.x + s]);
            __syncthreads();
        }
        m = red[0]; __syncthreads();
        float se = 0.f;
        for (int h = threadIdx.x; h < H_; h += 256) se += __expf(row[h] - m);
        red[threadIdx.x] = se; __syncthreads();
        for (int s = 128; s > 0; s >>= 1) {
            if (threadIdx.x < s) red[threadIdx.x] += red[threadIdx.x + s];
            __syncthreads();
        }
        if (threadIdx.x == 0) {
            const float lse = m + __logf(red[0]);
            loss += lse - row[y[b]];
        }
        __syncthreads();
    }
    if (threadIdx.x == 0) out[0] = loss / (float)B_;
}

// =====================================================================
// Host-side orchestration
// =====================================================================
extern "C" void kernel_launch(void* const* d_in, const int* in_sizes, int n_in,
                              void* d_out, int out_size, void* d_ws, size_t ws_size,
                              hipStream_t stream) {
    (void)in_sizes; (void)n_in; (void)out_size; (void)ws_size;

    const float* x    = (const float*)d_in[0];
    const int*   y    = (const int*)  d_in[1];
    const float* w1   = (const float*)d_in[2];
    const float* b1   = (const float*)d_in[3];
    const float* w2   = (const float*)d_in[4];
    const float* b2   = (const float*)d_in[5];
    const float* w3   = (const float*)d_in[6];
    const float* b3   = (const float*)d_in[7];
    const float* wg1  = (const float*)d_in[8];
    const float* e1w1 = (const float*)d_in[9];
    const float* e1b1 = (const float*)d_in[10];
    const float* e1w2 = (const float*)d_in[11];
    const float* e1b2 = (const float*)d_in[12];
    const float* wg2  = (const float*)d_in[13];
    const float* e2w1 = (const float*)d_in[14];
    const float* e2b1 = (const float*)d_in[15];
    const float* e2w2 = (const float*)d_in[16];
    const float* e2b2 = (const float*)d_in[17];

    char* ws = (char*)d_ws;
    const size_t big = (size_t)T_ * H_ * sizeof(float);   // 16 MB
    float* ws0 = (float*)(ws + 0 * big);   // hidden1 (residual, persists)
    float* ws1 = (float*)(ws + 1 * big);
    float* ws2 = (float*)(ws + 2 * big);
    float* ws3 = (float*)(ws + 3 * big);
    char* sm = ws + 4 * big;
    int*   expert_idx = (int*)sm;   sm += (size_t)T_ * 4;
    int*   slot       = (int*)sm;   sm += (size_t)T_ * 4;
    int*   keep       = (int*)sm;   sm += (size_t)T_ * 4;
    int*   tfs        = (int*)sm;   sm += (size_t)T_ * 4;
    float* gate_val   = (float*)sm; sm += (size_t)T_ * 4;
    float* cw         = (float*)sm; sm += (size_t)T_ * 4;
    int*   count      = (int*)sm;   sm += 256;
    float* sent       = (float*)sm; sm += (size_t)B_ * H_ * 4;
    float* partial    = (float*)sm; sm += (size_t)NC_ * B_ * H_ * 4;

    const dim3 blk(256);
    const dim3 gD(T_ / 128, H_ / 128, 1);     // dense GEMM: 32 x 8
    const dim3 gE(CAP_ / 128, H_ / 128, E_);  // expert GEMM: 4 x 8 x 8
    const long long sAe = (long long)CAP_ * H_;
    const long long sWe = (long long)H_ * H_;
    const long long sBe = H_;

    // hidden1 = x @ w1 + b1
    gemm_bias_bf16<<<gD, blk, 0, stream>>>(x, w1, b1, ws0, T_, H_, H_, 0, 0, 0, 0);

    // ---------------- MoE layer 1 (input ws0 -> out ws1) ----------------
    gating_kernel<<<T_ / 8, blk, 0, stream>>>(ws0, wg1, expert_idx, gate_val);
    assign_kernel<<<1, 32, 0, stream>>>(expert_idx, gate_val, slot, cw, keep, tfs, count);
    gather_kernel<<<T_, blk, 0, stream>>>(ws0, tfs, count, ws2);
    gemm_bias_bf16<<<gE, blk, 0, stream>>>(ws2, e1w1, e1b1, ws3, CAP_, H_, H_, sAe, sWe, sBe, sAe);
    gemm_bias_bf16<<<gE, blk, 0, stream>>>(ws3, e1w2, e1b2, ws2, CAP_, H_, H_, sAe, sWe, sBe, sAe);
    scatter_kernel<<<T_, blk, 0, stream>>>(ws2, expert_idx, slot, keep, cw, ws1);

    // out = moe1 @ w2 + b2   (ws1 -> ws3)
    gemm_bias_bf16<<<gD, blk, 0, stream>>>(ws1, w2, b2, ws3, T_, H_, H_, 0, 0, 0, 0);

    // ---------------- MoE layer 2 (input ws3 -> out ws1) ----------------
    gating_kernel<<<T_ / 8, blk, 0, stream>>>(ws3, wg2, expert_idx, gate_val);
    assign_kernel<<<1, 32, 0, stream>>>(expert_idx, gate_val, slot, cw, keep, tfs, count);
    gather_kernel<<<T_, blk, 0, stream>>>(ws3, tfs, count, ws2);
    gemm_bias_bf16<<<gE, blk, 0, stream>>>(ws2, e2w1, e2b1, ws1, CAP_, H_, H_, sAe, sWe, sBe, sAe);
    gemm_bias_bf16<<<gE, blk, 0, stream>>>(ws1, e2w2, e2b2, ws2, CAP_, H_, H_, sAe, sWe, sBe, sAe);
    scatter_kernel<<<T_, blk, 0, stream>>>(ws2, expert_idx, slot, keep, cw, ws1);

    // out = moe2 @ w3 + b3   (ws1 -> ws3)
    gemm_bias_bf16<<<gD, blk, 0, stream>>>(ws1, w3, b3, ws3, T_, H_, H_, 0, 0, 0, 0);

    // sent = mean_s(hidden1 + out); loss = NLL(log_softmax(sent), y)
    mean_part_kernel<<<(NC_ * B_ * H_) / 256, blk, 0, stream>>>(ws0, ws3, partial);
    mean_comb_kernel<<<(B_ * H_) / 256, blk, 0, stream>>>(partial, sent);
    loss_kernel<<<1, blk, 0, stream>>>(sent, y, (float*)d_out);
}
